// MultiheadAttention_81827716923513
// MI455X (gfx1250) — matrix operand order repaired and checked by reading, not compiled, not run on hardware
//
#include <hip/hip_runtime.h>
#include <hip/hip_bf16.h>
#include <math.h>

// B=8, N=1024, E=768, H=12, HD=64.  out = MHA(query) with qkv/out linear layers.
// All matmuls on v_wmma_f32_16x16x32_bf16; all tile staging via CDNA5
// global_load_async_to_lds_b128 with double-buffered LDS (ASYNCcnt pipelining).

typedef __attribute__((ext_vector_type(16))) __bf16 v16bf;
typedef __attribute__((ext_vector_type(8)))  float  v8f;

__device__ __forceinline__ unsigned short f2bf(float f) {
  union { float f; unsigned int u; } v; v.f = f;
  unsigned int u = v.u;
  u += 0x7fffu + ((u >> 16) & 1u);       // round-to-nearest-even
  return (unsigned short)(u >> 16);
}

__device__ __forceinline__ __bf16 bits2bf(unsigned short s) {
  union { unsigned short u; __bf16 b; } v; v.u = s;
  return v.b;
}

// CDNA5 async copy: 16 bytes global -> LDS per lane, tracked by ASYNCcnt.
__device__ __forceinline__ void async_b128(unsigned short* lds_dst,
                                           const unsigned short* gsrc) {
  const unsigned loff = (unsigned)(unsigned long long)lds_dst;
  asm volatile("global_load_async_to_lds_b128 %0, %1, off"
               :: "v"(loff), "v"(gsrc) : "memory");
}
__device__ __forceinline__ void wait_async() {
  asm volatile("s_wait_asynccnt 0x0" ::: "memory");
}

// A-fragment: 16x32 bf16 tile (row-major, leading dim ld; LDS or global).
// Lane m=lane&15 holds row m; half=lane>>4; VGPR i holds K = 2i + 8*half (+8 if i>=4).
__device__ __forceinline__ v16bf load_frag_a(const unsigned short* t, int ld) {
  const int lane = threadIdx.x & 31;
  const int m = lane & 15, hf = lane >> 4;
  const unsigned short* row = t + m * ld + 8 * hf;
  v16bf a;
#pragma unroll
  for (int i = 0; i < 8; ++i) {
    const int k = 2 * i + (i >= 4 ? 8 : 0);
    a[2 * i]     = bits2bf(row[k]);
    a[2 * i + 1] = bits2bf(row[k + 1]);
  }
  return a;
}

// B-fragment: 32x16 bf16 tile B[k][n] (k-major, leading dim ld).
// Lane n=lane&15 holds column n; half=lane>>4; VGPR i holds K = 2i + 8*half (+8 if i>=4).
__device__ __forceinline__ v16bf load_frag_b(const unsigned short* t, int ld) {
  const int lane = threadIdx.x & 31;
  const unsigned short* row = t + (lane & 15) + 8 * (lane >> 4) * ld;
  v16bf b;
#pragma unroll
  for (int i = 0; i < 8; ++i) {
    const int k = 2 * i + (i >= 4 ? 8 : 0);
    b[2 * i]     = bits2bf(row[k * ld]);
    b[2 * i + 1] = bits2bf(row[(k + 1) * ld]);
  }
  return b;
}

__device__ __forceinline__ v8f wmma_bf16(v16bf a, v16bf b, v8f c) {
  return __builtin_amdgcn_wmma_f32_16x16x32_bf16(false, a, false, b, (short)0, c,
                                                 false, false);
}

// ---------------------------------------------------------------------------
// Pre-pass A: f32 -> bf16, same layout (for query).
// ---------------------------------------------------------------------------
__global__ __launch_bounds__(256) void cvt_bf16_kernel(
    const float* __restrict__ src, unsigned short* __restrict__ dst, int n4) {
  const int i = blockIdx.x * 256 + threadIdx.x;
  if (i < n4) {
    const float4 f = reinterpret_cast<const float4*>(src)[i];
    ushort4 s;
    s.x = f2bf(f.x); s.y = f2bf(f.y); s.z = f2bf(f.z); s.w = f2bf(f.w);
    reinterpret_cast<ushort4*>(dst)[i] = s;
  }
}

// ---------------------------------------------------------------------------
// Pre-pass B: f32 [R][768] -> bf16 transposed [768][R] (64x64 tiles via LDS).
// ---------------------------------------------------------------------------
__global__ __launch_bounds__(256) void transpose_bf16_kernel(
    const float* __restrict__ src, unsigned short* __restrict__ dst, int R) {
  __shared__ __align__(16) unsigned short tile[64 * 68];
  const int t  = threadIdx.x;
  const int r0 = blockIdx.x * 64;    // row block in src
  const int c0 = blockIdx.y * 64;    // col block in src
#pragma unroll
  for (int it = 0; it < 4; ++it) {
    const int lin = t + 256 * it;    // 1024 quads
    const int r = lin >> 4, c4 = (lin & 15) * 4;
    const float4 f = *reinterpret_cast<const float4*>(
        &src[(size_t)(r0 + r) * 768 + c0 + c4]);
    ushort4 s;
    s.x = f2bf(f.x); s.y = f2bf(f.y); s.z = f2bf(f.z); s.w = f2bf(f.w);
    *reinterpret_cast<ushort4*>(&tile[r * 68 + c4]) = s;
  }
  __syncthreads();
#pragma unroll
  for (int it = 0; it < 4; ++it) {
    const int lin = t + 256 * it;
    const int c = lin >> 4, r4 = (lin & 15) * 4;
    ushort4 s;
    s.x = tile[(r4 + 0) * 68 + c];
    s.y = tile[(r4 + 1) * 68 + c];
    s.z = tile[(r4 + 2) * 68 + c];
    s.w = tile[(r4 + 3) * 68 + c];
    *reinterpret_cast<ushort4*>(&dst[(size_t)(c0 + c) * R + r0 + r4]) = s;
  }
}

// ---------------------------------------------------------------------------
// Kernel 1: qkv = x @ qkv_w^T + qkv_b ; scatter to Q (scaled), K^T, V (bf16).
// A from xb [8192,768] bf16, B from wtb [768,2304] bf16 (pre-transposed).
// Double-buffered async staging; 8 waves of 32x64; 24 k-steps.
// ---------------------------------------------------------------------------
__global__ __launch_bounds__(256) void qkv_gemm_kernel(
    const unsigned short* __restrict__ xb,   // [8192,768]
    const unsigned short* __restrict__ wtb,  // [768,2304]
    const float* __restrict__ bias,          // [2304]
    unsigned short* __restrict__ qb,         // [B,H,N,64] (pre-scaled 0.125)
    unsigned short* __restrict__ ktb,        // [B,H,64,N]
    unsigned short* __restrict__ vb)         // [B,H,N,64]
{
  __shared__ __align__(16) unsigned short As[2][128 * 40];
  __shared__ __align__(16) unsigned short Bs[2][32 * 136];

  const int t    = threadIdx.x;
  const int wave = t >> 5;
  const int lane = t & 31;
  const int hf   = lane >> 4;
  const int nidx = lane & 15;
  const int m0   = blockIdx.x * 128;
  const int j0   = blockIdx.y * 128;
  const int wm   = (wave & 3) * 32;
  const int wn   = (wave >> 2) * 64;

  auto stage = [&](int k0, int buf) {
#pragma unroll
    for (int it = 0; it < 2; ++it) {         // A: 128x32, 512 ops
      const int q = t + 256 * it;
      const int r = q >> 2, c8 = (q & 3) * 8;
      async_b128(&As[buf][r * 40 + c8], &xb[(size_t)(m0 + r) * 768 + k0 + c8]);
    }
#pragma unroll
    for (int it = 0; it < 2; ++it) {         // B: 32x128, 512 ops
      const int q = t + 256 * it;
      const int kk = q >> 4, c8 = (q & 15) * 8;
      async_b128(&Bs[buf][kk * 136 + c8],
                 &wtb[(size_t)(k0 + kk) * 2304 + j0 + c8]);
    }
  };

  v8f acc[2][4];
#pragma unroll
  for (int mi = 0; mi < 2; ++mi)
#pragma unroll
    for (int ni = 0; ni < 4; ++ni) { v8f z = {}; acc[mi][ni] = z; }

  stage(0, 0);
  for (int ks = 0; ks < 24; ++ks) {
    const int cur = ks & 1;
    wait_async();                  // own copies into buf[cur] complete
    __syncthreads();               // everyone done (and done reading buf[cur^1])
    if (ks + 1 < 24) stage((ks + 1) * 32, cur ^ 1);  // overlaps with compute

    v16bf af[2], bfr[4];
#pragma unroll
    for (int mi = 0; mi < 2; ++mi)
      af[mi] = load_frag_a(&As[cur][(wm + mi * 16) * 40], 40);
#pragma unroll
    for (int ni = 0; ni < 4; ++ni)
      bfr[ni] = load_frag_b(&Bs[cur][wn + ni * 16], 136);
#pragma unroll
    for (int mi = 0; mi < 2; ++mi)
#pragma unroll
      for (int ni = 0; ni < 4; ++ni)
        acc[mi][ni] = wmma_bf16(af[mi], bfr[ni], acc[mi][ni]);
  }

  // Epilogue: sel is uniform per block (768 % 128 == 0)
  const int sel = j0 / 768;                // 0=Q, 1=K, 2=V (scalar)
  const int e0  = j0 - sel * 768 + wn;     // head-space column base
#pragma unroll
  for (int mi = 0; mi < 2; ++mi)
#pragma unroll
    for (int ni = 0; ni < 4; ++ni)
#pragma unroll
      for (int r = 0; r < 8; ++r) {
        const int row = m0 + wm + mi * 16 + r + 8 * hf;  // bn
        const int col = j0 + wn + ni * 16 + nidx;
        const float v = acc[mi][ni][r] + bias[col];
        const int b = row >> 10, n = row & 1023;
        const int e = e0 + ni * 16 + nidx;
        const int h = e >> 6, d = e & 63;
        const int bh = b * 12 + h;
        if (sel == 0)
          qb[((size_t)bh * 1024 + n) * 64 + d] = f2bf(v * 0.125f);
        else if (sel == 1)
          ktb[((size_t)bh * 64 + d) * 1024 + n] = f2bf(v);
        else
          vb[((size_t)bh * 1024 + n) * 64 + d] = f2bf(v);
      }
}

// ---------------------------------------------------------------------------
// Kernel 2: flash-style attention per (b,h). 128 query rows/block, 8 waves of
// 16 rows; Q frags direct from global; 64-key K/V tiles double-buffered async.
// ---------------------------------------------------------------------------
__global__ __launch_bounds__(256) void attn_kernel(
    const unsigned short* __restrict__ qb,   // [B,H,N,64]
    const unsigned short* __restrict__ ktb,  // [B,H,64,N]
    const unsigned short* __restrict__ vb,   // [B,H,N,64]
    unsigned short* __restrict__ ctx)        // [B,N,768]
{
  __shared__ __align__(16) unsigned short Ks[2][64 * 72];   // Kt[d][k]
  __shared__ __align__(16) unsigned short Vs[2][64 * 72];   // V[k][d]
  __shared__ __align__(16) unsigned short Ps[8][16 * 72];   // per-wave P scratch

  const int t    = threadIdx.x;
  const int wave = t >> 5;
  const int lane = t & 31;
  const int hf   = lane >> 4;
  const int nidx = lane & 15;
  const int q0   = blockIdx.x * 128;
  const int bh   = blockIdx.y;               // 0..95
  const int b    = bh / 12, h = bh % 12;

  const unsigned short* Qg = qb  + (size_t)bh * 1024 * 64;
  const unsigned short* Kg = ktb + (size_t)bh * 64 * 1024;
  const unsigned short* Vg = vb  + (size_t)bh * 1024 * 64;

  auto stage_kv = [&](int k0, int buf) {
#pragma unroll
    for (int it = 0; it < 2; ++it) {
      const int q = t + 256 * it;            // 512 ops each
      const int r = q >> 3, c8 = (q & 7) * 8;
      async_b128(&Ks[buf][r * 72 + c8], &Kg[(size_t)r * 1024 + k0 + c8]);
      async_b128(&Vs[buf][r * 72 + c8], &Vg[(size_t)(k0 + r) * 64 + c8]);
    }
  };

  // Q fragments straight from global (row-major, each row used by one wave)
  const v16bf aq0 = load_frag_a(&Qg[(size_t)(q0 + wave * 16) * 64], 64);
  const v16bf aq1 = load_frag_a(&Qg[(size_t)(q0 + wave * 16) * 64 + 32], 64);

  v8f o[4];
#pragma unroll
  for (int dt = 0; dt < 4; ++dt) { v8f z = {}; o[dt] = z; }
  float mrow[8], lrow[8];
#pragma unroll
  for (int r = 0; r < 8; ++r) { mrow[r] = -3.0e38f; lrow[r] = 0.0f; }

  stage_kv(0, 0);
  for (int kb = 0; kb < 16; ++kb) {
    const int cur = kb & 1;
    wait_async();
    __syncthreads();
    if (kb + 1 < 16) stage_kv((kb + 1) * 64, cur ^ 1);  // overlaps compute

    // S = Q K^T for 64 keys: 4 column tiles x (2 d-chunks)
    v8f s[4];
#pragma unroll
    for (int nt = 0; nt < 4; ++nt) {
      v16bf b0 = load_frag_b(&Ks[cur][nt * 16], 72);
      v16bf b1 = load_frag_b(&Ks[cur][32 * 72 + nt * 16], 72);
      v8f z = {};
      z = wmma_bf16(aq0, b0, z);
      s[nt] = wmma_bf16(aq1, b1, z);
    }

    // online softmax (row M = r + 8*hf lives in the 16 lanes of its half)
    float mnew[8];
#pragma unroll
    for (int r = 0; r < 8; ++r) {
      float mx = fmaxf(fmaxf(s[0][r], s[1][r]), fmaxf(s[2][r], s[3][r]));
      mx = fmaxf(mx, __shfl_xor(mx, 1, 32));
      mx = fmaxf(mx, __shfl_xor(mx, 2, 32));
      mx = fmaxf(mx, __shfl_xor(mx, 4, 32));
      mx = fmaxf(mx, __shfl_xor(mx, 8, 32));
      mnew[r] = fmaxf(mrow[r], mx);
    }
#pragma unroll
    for (int nt = 0; nt < 4; ++nt)
#pragma unroll
      for (int r = 0; r < 8; ++r)
        s[nt][r] = __expf(s[nt][r] - mnew[r]);
#pragma unroll
    for (int r = 0; r < 8; ++r) {
      float sum = s[0][r] + s[1][r] + s[2][r] + s[3][r];
      sum += __shfl_xor(sum, 1, 32);
      sum += __shfl_xor(sum, 2, 32);
      sum += __shfl_xor(sum, 4, 32);
      sum += __shfl_xor(sum, 8, 32);
      const float alpha = __expf(mrow[r] - mnew[r]);
      lrow[r] = lrow[r] * alpha + sum;
      mrow[r] = mnew[r];
#pragma unroll
      for (int dt = 0; dt < 4; ++dt) o[dt][r] *= alpha;
    }

    // P: C-frag layout -> LDS -> reload as A-frags
    unsigned short* Pw = Ps[wave];
#pragma unroll
    for (int nt = 0; nt < 4; ++nt)
#pragma unroll
      for (int r = 0; r < 8; ++r)
        Pw[(r + 8 * hf) * 72 + nt * 16 + nidx] = f2bf(s[nt][r]);
    __syncthreads();

    v16bf ap0 = load_frag_a(&Pw[0], 72);    // keys k0..k0+31
    v16bf ap1 = load_frag_a(&Pw[32], 72);   // keys k0+32..k0+63
#pragma unroll
    for (int dt = 0; dt < 4; ++dt) {
      o[dt] = wmma_bf16(ap0, load_frag_b(&Vs[cur][dt * 16], 72), o[dt]);
      o[dt] = wmma_bf16(ap1, load_frag_b(&Vs[cur][32 * 72 + dt * 16], 72), o[dt]);
    }
  }

  // finalize: O / l -> ctx[b, n, h*64 + d] (bf16)
#pragma unroll
  for (int r = 0; r < 8; ++r) {
    const int n = q0 + wave * 16 + r + 8 * hf;
    const float inv = 1.0f / lrow[r];
    unsigned short* dst = &ctx[((size_t)(b * 1024 + n)) * 768 + h * 64];
#pragma unroll
    for (int dt = 0; dt < 4; ++dt)
      dst[dt * 16 + nidx] = f2bf(o[dt][r] * inv);
  }
}

// ---------------------------------------------------------------------------
// Kernel 3: out = ctx @ out_w^T + out_b  (M=8192, K=768, N=768), f32 output.
// A from ctx bf16, B from owtb [768][768] bf16 (pre-transposed).
// ---------------------------------------------------------------------------
__global__ __launch_bounds__(256) void out_gemm_kernel(
    const unsigned short* __restrict__ ctx,   // [8192, 768]
    const unsigned short* __restrict__ owtb,  // [768, 768]
    const float* __restrict__ bias,           // [768]
    float* __restrict__ out)                  // [8192, 768]
{
  __shared__ __align__(16) unsigned short As[2][128 * 40];
  __shared__ __align__(16) unsigned short Bs[2][32 * 136];

  const int t    = threadIdx.x;
  const int wave = t >> 5;
  const int lane = t & 31;
  const int hf   = lane >> 4;
  const int nidx = lane & 15;
  const int m0   = blockIdx.x * 128;
  const int j0   = blockIdx.y * 128;
  const int wm   = (wave & 3) * 32;
  const int wn   = (wave >> 2) * 64;

  auto stage = [&](int k0, int buf) {
#pragma unroll
    for (int it = 0; it < 2; ++it) {
      const int q = t + 256 * it;
      const int r = q >> 2, c8 = (q & 3) * 8;
      async_b128(&As[buf][r * 40 + c8], &ctx[(size_t)(m0 + r) * 768 + k0 + c8]);
    }
#pragma unroll
    for (int it = 0; it < 2; ++it) {
      const int q = t + 256 * it;
      const int kk = q >> 4, c8 = (q & 15) * 8;
      async_b128(&Bs[buf][kk * 136 + c8],
                 &owtb[(size_t)(k0 + kk) * 768 + j0 + c8]);
    }
  };

  v8f acc[2][4];
#pragma unroll
  for (int mi = 0; mi < 2; ++mi)
#pragma unroll
    for (int ni = 0; ni < 4; ++ni) { v8f z = {}; acc[mi][ni] = z; }

  stage(0, 0);
  for (int ks = 0; ks < 24; ++ks) {
    const int cur = ks & 1;
    wait_async();
    __syncthreads();
    if (ks + 1 < 24) stage((ks + 1) * 32, cur ^ 1);

    v16bf af[2], bfr[4];
#pragma unroll
    for (int mi = 0; mi < 2; ++mi)
      af[mi] = load_frag_a(&As[cur][(wm + mi * 16) * 40], 40);
#pragma unroll
    for (int ni = 0; ni < 4; ++ni)
      bfr[ni] = load_frag_b(&Bs[cur][wn + ni * 16], 136);
#pragma unroll
    for (int mi = 0; mi < 2; ++mi)
#pragma unroll
      for (int ni = 0; ni < 4; ++ni)
        acc[mi][ni] = wmma_bf16(af[mi], bfr[ni], acc[mi][ni]);
  }

#pragma unroll
  for (int mi = 0; mi < 2; ++mi)
#pragma unroll
    for (int ni = 0; ni < 4; ++ni)
#pragma unroll
      for (int r = 0; r < 8; ++r) {
        const int row = m0 + wm + mi * 16 + r + 8 * hf;
        const int col = j0 + wn + ni * 16 + nidx;
        out[(size_t)row * 768 + col] = acc[mi][ni][r] + bias[col];
      }
}

// ---------------------------------------------------------------------------
extern "C" void kernel_launch(void* const* d_in, const int* in_sizes, int n_in,
                              void* d_out, int out_size, void* d_ws, size_t ws_size,
                              hipStream_t stream) {
  (void)in_sizes; (void)n_in; (void)out_size; (void)ws_size;
  const float* query = (const float*)d_in[0];  // [8,1024,768]
  const float* qkv_w = (const float*)d_in[1];  // [2304,768]
  const float* qkv_b = (const float*)d_in[2];  // [2304]
  const float* out_w = (const float*)d_in[3];  // [768,768]
  const float* out_b = (const float*)d_in[4];  // [768]
  float* out = (float*)d_out;                  // [8,1024,768]

  const size_t HBUF = (size_t)8 * 12 * 1024 * 64;   // 6,291,456
  unsigned short* xb   = (unsigned short*)d_ws;     // [8192,768]
  unsigned short* wtb  = xb + HBUF;                 // [768,2304]
  unsigned short* owtb = wtb + (size_t)768 * 2304;  // [768,768]
  unsigned short* qb   = owtb + (size_t)768 * 768;
  unsigned short* ktb  = qb + HBUF;
  unsigned short* vb   = ktb + HBUF;
  unsigned short* ctx  = vb + HBUF;                 // [8192,768]
  // total ws use ~67.6 MB

  cvt_bf16_kernel<<<6144, 256, 0, stream>>>(query, xb, 1572864);
  transpose_bf16_kernel<<<dim3(36, 12), 256, 0, stream>>>(qkv_w, wtb, 2304);
  transpose_bf16_kernel<<<dim3(12, 12), 256, 0, stream>>>(out_w, owtb, 768);

  qkv_gemm_kernel<<<dim3(64, 18), 256, 0, stream>>>(xb, wtb, qkv_b, qb, ktb, vb);
  attn_kernel<<<dim3(8, 96), 256, 0, stream>>>(qb, ktb, vb, ctx);
  out_gemm_kernel<<<dim3(64, 6), 256, 0, stream>>>(ctx, owtb, out_b, out);
}
